// ShiftedWindowAttention3d_22144851378723
// MI455X (gfx1250) — matrix operand order repaired and checked by reading, not compiled, not run on hardware
//
#include <hip/hip_runtime.h>

typedef __attribute__((ext_vector_type(16))) __bf16 v16bf;
typedef __attribute__((ext_vector_type(8)))  __bf16 v8bf;
typedef __attribute__((ext_vector_type(8)))  float  v8f;
typedef __attribute__((ext_vector_type(4)))  float  v4f;

namespace {
constexpr int kC = 128;
constexpr int kHeads = 4, kHd = 32, kVol = 392;
constexpr int kQPad = 400, kKPad = 416;
constexpr float kScale = 0.17677669529663687f; // 1/sqrt(32)
}

// ---------------------------------------------------------------------------
// Fragment helpers (layouts per CDNA5 ISA 7.12.2)
// A-matrix 16x32 bf16: lane L holds row M=L%16; lanes 0-15 hold K = 0..7 and
// 16..23 (base 0), lanes 16-31 hold K = 8..15 and 24..31 (base 8).
__device__ inline v16bf load_afrag_bf(const __bf16* p) {
  v8bf lo = *(const v8bf*)p;
  v8bf hi = *(const v8bf*)(p + 16);
  v16bf r;
#pragma unroll
  for (int i = 0; i < 8; ++i) { r[i] = lo[i]; r[i + 8] = hi[i]; }
  return r;
}
__device__ inline v16bf cvt_afrag_f32(const float* p) {
  v4f a0 = *(const v4f*)(p);
  v4f a1 = *(const v4f*)(p + 4);
  v4f a2 = *(const v4f*)(p + 16);
  v4f a3 = *(const v4f*)(p + 20);
  v16bf r;
#pragma unroll
  for (int i = 0; i < 4; ++i) {
    r[i]      = (__bf16)a0[i];
    r[4 + i]  = (__bf16)a1[i];
    r[8 + i]  = (__bf16)a2[i];
    r[12 + i] = (__bf16)a3[i];
  }
  return r;
}
// B-matrix 32x16 bf16: lane L holds column N=L%16; same K split as A:
// K = base..base+7 and base+16..base+23, base = (L<16 ? 0 : 8).
__device__ inline v16bf load_bfrag_bf(const __bf16* p) {
  v8bf lo = *(const v8bf*)p;
  v8bf hi = *(const v8bf*)(p + 16);
  v16bf r;
#pragma unroll
  for (int i = 0; i < 8; ++i) { r[i] = lo[i]; r[i + 8] = hi[i]; }
  return r;
}
__device__ inline v16bf cvt_bfrag_f32(const float* p) {
  v16bf r;
#pragma unroll
  for (int j = 0; j < 4; ++j) {
    v4f a = *(const v4f*)(p + 4 * (j & 1) + 16 * (j >> 1));
#pragma unroll
    for (int i = 0; i < 4; ++i) r[4 * j + i] = (__bf16)a[i];
  }
  return r;
}

__device__ inline v8f wmma_bf16(v16bf a, v16bf b, v8f c) {
  return __builtin_amdgcn_wmma_f32_16x16x32_bf16(false, a, false, b, (short)0,
                                                 c, false, false);
}

// ---------------------------------------------------------------------------
// Zero the padded tail rows of Q (rows 392..399), K and V^T (rows 392..415).
__global__ void swin_pad_init(__bf16* Q, __bf16* K, __bf16* V) {
  int idx = blockIdx.x * blockDim.x + threadIdx.x;  // < 786432
  if (idx < 262144) {                 // 1024 groups * 8 rows * 32 dims
    int g = idx >> 8;
    int rrem = idx & 255;
    int qi = kVol + (rrem >> 5);
    int dim = rrem & 31;
    Q[((size_t)g * kQPad + qi) * kHd + dim] = (__bf16)0.f;
  }
  if (idx < 786432) {                 // 1024 groups * 24 rows * 32 dims
    int g = idx / 768;
    int rrem = idx - g * 768;
    int qi = kVol + rrem / 32;
    int dim = rrem & 31;
    K[((size_t)g * kKPad + qi) * kHd + dim] = (__bf16)0.f;
    int d2 = rrem / 24;               // 0..31
    int r2 = rrem - d2 * 24;          // 0..23
    V[((size_t)g * kHd + d2) * kKPad + kVol + r2] = (__bf16)0.f;
  }
}

// ---------------------------------------------------------------------------
// Stage 1: fused shift + window partition gather + QKV GEMM (bf16 WMMA).
// One wave per 16x16 output tile; K=128 in 4 steps of 32.
__global__ __launch_bounds__(32) void swin_qkv_kernel(
    const float* __restrict__ x, const float* __restrict__ wqkv,
    const float* __restrict__ bqkv, __bf16* __restrict__ Q,
    __bf16* __restrict__ K, __bf16* __restrict__ V) {
  const int lane = threadIdx.x;
  const int lm = lane & 15, hi = lane >> 4;
  const int mt = blockIdx.x, nt = blockIdx.y;
  const int rowA = mt * 16 + lm;
  const int col = nt * 16 + lm;

  int bw = rowA / kVol, r = rowA - bw * kVol;
  int b = bw >> 7, widx = bw & 127;
  int it = widx >> 6, ih = (widx >> 3) & 7, iw = widx & 7;
  int lt = r / 49, rem = r - lt * 49, lh = rem / 7, lw = rem - lh * 7;
  int t = (it * 8 + lt + 4) & 15;                       // roll -SHIFT gather
  int h = ih * 7 + lh + 3; if (h >= 56) h -= 56;
  int w = iw * 7 + lw + 3; if (w >= 56) w -= 56;
  const float* xrow = x + (((size_t)(b * 16 + t) * 56 + h) * 56 + w) * kC;
  const float* wrow = wqkv + (size_t)col * kC;
  const int akb = hi * 8, bkb = hi * 8;

  v8f acc = {};
#pragma unroll
  for (int ks = 0; ks < 4; ++ks) {
    v16bf a  = cvt_afrag_f32(xrow + ks * 32 + akb);
    v16bf bb = cvt_bfrag_f32(wrow + ks * 32 + bkb);
    acc = wmma_bf16(a, bb, acc);
  }

  const float bias = bqkv[col];
#pragma unroll
  for (int d = 0; d < 8; ++d) {
    int m = mt * 16 + hi * 8 + d;
    int bw2 = m / kVol, qi = m - bw2 * kVol;
    float val = acc[d] + bias;
    if (nt < 8) {                                       // Q (pre-scaled)
      int head = col >> 5, dim = col & 31;
      Q[(((size_t)bw2 * kHeads + head) * kQPad + qi) * kHd + dim] =
          (__bf16)(val * kScale);
    } else if (nt < 16) {                               // K row-major
      int nn = col - 128; int head = nn >> 5, dim = nn & 31;
      K[(((size_t)bw2 * kHeads + head) * kKPad + qi) * kHd + dim] = (__bf16)val;
    } else {                                            // V transposed
      int nn = col - 256; int head = nn >> 5, dim = nn & 31;
      V[(((size_t)bw2 * kHeads + head) * kHd + dim) * kKPad + qi] = (__bf16)val;
    }
  }
}

// ---------------------------------------------------------------------------
// Stage 2: flash-style window attention. One wave per (16 q-rows, bw, head).
// 13 chunks of 32 keys: S=Q*K^T (2 WMMA), bias+mask, online softmax,
// P transposed via LDS, O += P*V (2 WMMA).
__global__ __launch_bounds__(32) void swin_attn_kernel(
    const __bf16* __restrict__ Q, const __bf16* __restrict__ K,
    const __bf16* __restrict__ V, const float* __restrict__ rtab,
    __bf16* __restrict__ AO) {
  __shared__ float Pl[16][32];
  const int lane = threadIdx.x;
  const int lm = lane & 15, hi = lane >> 4;
  const int qt = blockIdx.x;
  const int gh = blockIdx.y;
  const int bw = gh >> 2, head = gh & 3;
  const int widx = bw & 127;
  const int it = widx >> 6, ih = (widx >> 3) & 7, iw = widx & 7;

  const __bf16* Qb = Q + ((size_t)bw * kHeads + head) * kQPad * kHd;
  const __bf16* Kb = K + ((size_t)bw * kHeads + head) * kKPad * kHd;
  const __bf16* Vb = V + ((size_t)bw * kHeads + head) * kHd * kKPad;

  const int akb = hi * 8, bkb = hi * 8;
  v16bf qa = load_afrag_bf(Qb + (qt * 16 + lm) * kHd + akb);

  // query-side window coords + shift-mask region per accumulator row
  int qlt[8], qlh[8], qlw[8], qcnt[8];
  bool qv[8];
#pragma unroll
  for (int d = 0; d < 8; ++d) {
    int qi = qt * 16 + hi * 8 + d;
    qv[d] = qi < kVol;
    int qq = qv[d] ? qi : 0;
    int a = qq / 49, rem = qq - a * 49, b2 = rem / 7, c2 = rem - b2 * 7;
    qlt[d] = a; qlh[d] = b2; qlw[d] = c2;
    int t = it * 8 + a, hh = ih * 7 + b2, ww = iw * 7 + c2;
    int rt = (t < 8) ? 0 : ((t < 12) ? 1 : 2);
    int rh = (hh < 49) ? 0 : ((hh < 53) ? 1 : 2);
    int rw = (ww < 49) ? 0 : ((ww < 53) ? 1 : 2);
    qcnt[d] = rt * 9 + rh * 3 + rw;
  }

  float mrun[8], lrun[8];
  v8f o0 = {}, o1 = {};
#pragma unroll
  for (int d = 0; d < 8; ++d) { mrun[d] = -1e30f; lrun[d] = 0.f; }

  for (int c = 0; c < 13; ++c) {
    const int cb = c * 32;
    v8f s0 = {}, s1 = {};
    {
      v16bf kf0 = load_bfrag_bf(Kb + (size_t)(cb + lm) * kHd + bkb);
      v16bf kf1 = load_bfrag_bf(Kb + (size_t)(cb + 16 + lm) * kHd + bkb);
      s0 = wmma_bf16(qa, kf0, s0);
      s1 = wmma_bf16(qa, kf1, s1);
    }
    // relative-position bias + shift mask + pad masking
#pragma unroll
    for (int j = 0; j < 2; ++j) {
      v8f& s = j ? s1 : s0;
      int key = cb + j * 16 + lm;
      bool kvld = key < kVol;
      int kk = kvld ? key : 0;
      int a = kk / 49, rem = kk - a * 49, b2 = rem / 7, c2 = rem - b2 * 7;
      int t = it * 8 + a, hh = ih * 7 + b2, ww = iw * 7 + c2;
      int rt = (t < 8) ? 0 : ((t < 12) ? 1 : 2);
      int rh = (hh < 49) ? 0 : ((hh < 53) ? 1 : 2);
      int rw = (ww < 49) ? 0 : ((ww < 53) ? 1 : 2);
      int kcnt = rt * 9 + rh * 3 + rw;
#pragma unroll
      for (int d = 0; d < 8; ++d) {
        if (kvld && qv[d]) {
          int idx = (qlt[d] - a + 7) * 169 + (qlh[d] - b2 + 6) * 13 +
                    (qlw[d] - c2 + 6);
          float add = rtab[idx * kHeads + head];
          if (qcnt[d] != kcnt) add -= 100.f;
          s[d] += add;
        } else {
          s[d] = -1e30f;
        }
      }
    }
    // online softmax (row reductions across the 16-lane group)
    float cm[8], p0[8], p1[8], ps[8];
#pragma unroll
    for (int d = 0; d < 8; ++d) cm[d] = fmaxf(s0[d], s1[d]);
#pragma unroll
    for (int off = 1; off < 16; off <<= 1)
#pragma unroll
      for (int d = 0; d < 8; ++d)
        cm[d] = fmaxf(cm[d], __shfl_xor(cm[d], off, 32));
#pragma unroll
    for (int d = 0; d < 8; ++d) {
      float mn = fmaxf(mrun[d], cm[d]);
      float sc = __expf(mrun[d] - mn);
      mrun[d] = mn;
      lrun[d] *= sc;
      o0[d] *= sc; o1[d] *= sc;
      p0[d] = __expf(s0[d] - mn);
      p1[d] = __expf(s1[d] - mn);
      ps[d] = p0[d] + p1[d];
    }
#pragma unroll
    for (int off = 1; off < 16; off <<= 1)
#pragma unroll
      for (int d = 0; d < 8; ++d) ps[d] += __shfl_xor(ps[d], off, 32);
#pragma unroll
    for (int d = 0; d < 8; ++d) lrun[d] += ps[d];

    // transpose P (C layout -> A layout) through LDS
    __syncthreads();
#pragma unroll
    for (int d = 0; d < 8; ++d) {
      Pl[hi * 8 + d][lm] = p0[d];
      Pl[hi * 8 + d][lm + 16] = p1[d];
    }
    __syncthreads();
    v16bf pa;
#pragma unroll
    for (int i = 0; i < 8; ++i) {
      pa[i]     = (__bf16)Pl[lm][akb + i];
      pa[i + 8] = (__bf16)Pl[lm][akb + 16 + i];
    }
    v16bf vf0 = load_bfrag_bf(Vb + (size_t)(lm) * kKPad + cb + bkb);
    v16bf vf1 = load_bfrag_bf(Vb + (size_t)(16 + lm) * kKPad + cb + bkb);
    o0 = wmma_bf16(pa, vf0, o0);
    o1 = wmma_bf16(pa, vf1, o1);
  }

#pragma unroll
  for (int d = 0; d < 8; ++d) {
    int qi = qt * 16 + hi * 8 + d;
    if (qi < kVol) {
      float inv = 1.0f / lrun[d];
      size_t base = ((size_t)bw * kVol + qi) * kC + head * kHd;
      AO[base + lm]      = (__bf16)(o0[d] * inv);
      AO[base + 16 + lm] = (__bf16)(o1[d] * inv);
    }
  }
}

// ---------------------------------------------------------------------------
// Stage 3: projection GEMM + window reverse + roll-back scatter into d_out.
__global__ __launch_bounds__(32) void swin_proj_kernel(
    const __bf16* __restrict__ AO, const float* __restrict__ wproj,
    const float* __restrict__ bproj, float* __restrict__ out) {
  const int lane = threadIdx.x;
  const int lm = lane & 15, hi = lane >> 4;
  const int mt = blockIdx.x, nt = blockIdx.y;
  const int rowA = mt * 16 + lm;
  const int col = nt * 16 + lm;
  const int akb = hi * 8, bkb = hi * 8;
  const __bf16* arow = AO + (size_t)rowA * kC;
  const float* wrow = wproj + (size_t)col * kC;

  v8f acc = {};
#pragma unroll
  for (int ks = 0; ks < 4; ++ks) {
    v16bf a  = load_afrag_bf(arow + ks * 32 + akb);
    v16bf bb = cvt_bfrag_f32(wrow + ks * 32 + bkb);
    acc = wmma_bf16(a, bb, acc);
  }

  const float bias = bproj[col];
#pragma unroll
  for (int d = 0; d < 8; ++d) {
    int m = mt * 16 + hi * 8 + d;
    int bw = m / kVol, r = m - bw * kVol;
    int b = bw >> 7, widx = bw & 127;
    int it = widx >> 6, ih = (widx >> 3) & 7, iw = widx & 7;
    int lt = r / 49, rem = r - lt * 49, lh = rem / 7, lw = rem - lh * 7;
    int t = (it * 8 + lt + 4) & 15;                     // roll +SHIFT scatter
    int h = ih * 7 + lh + 3; if (h >= 56) h -= 56;
    int w = iw * 7 + lw + 3; if (w >= 56) w -= 56;
    out[(((size_t)(b * 16 + t) * 56 + h) * 56 + w) * kC + col] = acc[d] + bias;
  }
}

// ---------------------------------------------------------------------------
extern "C" void kernel_launch(void* const* d_in, const int* in_sizes, int n_in,
                              void* d_out, int out_size, void* d_ws,
                              size_t ws_size, hipStream_t stream) {
  const float* x     = (const float*)d_in[0];
  const float* wqkv  = (const float*)d_in[1];
  const float* bqkv  = (const float*)d_in[2];
  const float* wproj = (const float*)d_in[3];
  const float* bproj = (const float*)d_in[4];
  const float* rtab  = (const float*)d_in[5];
  float* out = (float*)d_out;

  char* ws = (char*)d_ws;
  // bf16 staging buffers (group = bw*4+head, 1024 groups)
  const size_t oQ = 0;                       // 1024*400*32*2 = 26,214,400 B
  const size_t oK = oQ + (size_t)1024 * 400 * 32 * 2;
  const size_t oV = oK + (size_t)1024 * 416 * 32 * 2;   // 27,262,976 B each
  const size_t oA = oV + (size_t)1024 * 416 * 32 * 2;   // AO: 256*392*128 bf16
  __bf16* Q  = (__bf16*)(ws + oQ);
  __bf16* K  = (__bf16*)(ws + oK);
  __bf16* V  = (__bf16*)(ws + oV);
  __bf16* AO = (__bf16*)(ws + oA);

  swin_pad_init<<<3072, 256, 0, stream>>>(Q, K, V);

  dim3 gA(6272, 24);   // 100352/16 M-tiles x 384/16 N-tiles
  swin_qkv_kernel<<<gA, 32, 0, stream>>>(x, wqkv, bqkv, Q, K, V);

  dim3 gB(25, 1024);   // 25 q-tiles x (256 windows * 4 heads)
  swin_attn_kernel<<<gB, 32, 0, stream>>>(Q, K, V, rtab, AO);

  dim3 gC(6272, 8);    // 100352/16 x 128/16
  swin_proj_kernel<<<gC, 32, 0, stream>>>(AO, wproj, bproj, out);
}
